// MLAAttention_20134806684223
// MI455X (gfx1250) — compile-verified, matrix operand order checked by reading, not run on hardware
//
#include <hip/hip_runtime.h>
#include <hip/hip_bf16.h>

typedef __attribute__((ext_vector_type(16))) __bf16 v16bf_t;
typedef __attribute__((ext_vector_type(8)))  __bf16 v8bf_t;
typedef __attribute__((ext_vector_type(4)))  __bf16 v4bf_t;
typedef __attribute__((ext_vector_type(8)))  float  v8f_t;
typedef __attribute__((ext_vector_type(4)))  float  v4f_t;
typedef __attribute__((ext_vector_type(16))) float  v16f_t;
typedef __attribute__((ext_vector_type(4)))  int    v4i_t;

#if defined(__has_builtin)
#if __has_builtin(__builtin_amdgcn_global_load_async_to_lds_b128) && \
    __has_builtin(__builtin_amdgcn_s_wait_asynccnt)
#define USE_ASYNC_LDS 1
#endif
#endif

// Native conversions -> v_cvt_pk_bf16_f32 (RTNE in hardware)
__device__ __forceinline__ __bf16 f2bf(float x) { return (__bf16)x; }
__device__ __forceinline__ v4bf_t cvt4(v4f_t v) {
    return __builtin_convertvector(v, v4bf_t);
}

// Load one 16x32 bf16 A/B fragment row for this lane from a 32-element,
// 16B-aligned LDS row at p. Under the ISA K-mapping the lane's 16 elements
// are exactly {p[half*8 .. +7], p[16+half*8 .. +7]} in order.
__device__ __forceinline__ v16bf_t load_frag(const __bf16* p, int half) {
    const v8bf_t lo = *(const v8bf_t*)(p + half * 8);
    const v8bf_t hi = *(const v8bf_t*)(p + 16 + half * 8);
    return __builtin_shufflevector(lo, hi, 0, 1, 2, 3, 4, 5, 6, 7,
                                           8, 9, 10, 11, 12, 13, 14, 15);
}

// ---------------------------------------------------------------------------
// One-time weight prep: W[K][N] fp32 -> Wt[N][K] bf16 (LDS-tiled transpose).
// K, N multiples of 32. 256 threads, 32x32 tiles.
// ---------------------------------------------------------------------------
__global__ __launch_bounds__(256) void transpose_cvt_kernel(
    const float* __restrict__ W, __bf16* __restrict__ Wt, int K, int N)
{
    __shared__ float tile[32][33];
    const int kb = blockIdx.y * 32, nb = blockIdx.x * 32;
    const int tx = threadIdx.x & 31, ty = threadIdx.x >> 5;   // ty = 0..7
    #pragma unroll
    for (int i = 0; i < 32; i += 8)
        tile[ty + i][tx] = W[(size_t)(kb + ty + i) * N + nb + tx];
    __syncthreads();
    #pragma unroll
    for (int i = 0; i < 32; i += 8)
        Wt[(size_t)(nb + ty + i) * K + kb + tx] = f2bf(tile[tx][ty + i]);
}

// ---------------------------------------------------------------------------
// Tiled GEMM: C[M,N] = A[M,K] @ B[K,N]; A fp32, B pre-transposed bf16 Bt[N][K].
// Block tile 256x64, K step 32, 256 threads = 8 waves, each wave 32 rows
// (2 A-frags reused over 4 B-frags -> 8 WMMA per K step per wave).
// B tile staged with one global_load_async_to_lds_b128 per lane when the
// builtin is available (Bt layout [n][k] == fragment layout, no transpose).
// Requires M%256==0, N%64==0, K%32==0 (true for all uses here).
// ---------------------------------------------------------------------------
__global__ __launch_bounds__(256) void gemm_bf16_wmma(
    const float* __restrict__ A, const __bf16* __restrict__ Bt,
    float* __restrict__ C, int M, int N, int K)
{
    __shared__ __bf16 As[256 * 32];   // [m][k]
    __shared__ __bf16 Bs[64 * 32];    // [n][k]

    const int tid  = threadIdx.x;
    const int lane = tid & 31;
    const int wave = tid >> 5;
    const int half = (lane >> 4) & 1;
    const int ln   = lane & 15;

    const int n0 = blockIdx.x * 64;
    const int m0 = blockIdx.y * 256;

    v8f_t acc[2][4];
    #pragma unroll
    for (int g = 0; g < 2; ++g)
        #pragma unroll
        for (int t = 0; t < 4; ++t)
            #pragma unroll
            for (int r = 0; r < 8; ++r) acc[g][t][r] = 0.0f;

    // B-tile copy: 64 rows x 32 k = 256 chunks of 8 bf16 (16B); 1 per thread.
    const int bn = tid >> 2, bk = (tid & 3) * 8;

    for (int k0 = 0; k0 < K; k0 += 32) {
        if (k0) __syncthreads();
        // A tile 256x32 fp32 -> bf16: 2048 float4 loads, 8 per thread
        #pragma unroll
        for (int e = 0; e < 8; ++e) {
            int idx4 = e * 256 + tid;
            int r = idx4 >> 3, c = (idx4 & 7) * 4;
            v4f_t v = *(const v4f_t*)(A + (size_t)(m0 + r) * K + k0 + c);
            *(v4bf_t*)(&As[r * 32 + c]) = cvt4(v);
        }
        // B tile: straight bf16 block copy, fragment-ready layout
        {
            const __bf16* g = Bt + (size_t)(n0 + bn) * K + k0 + bk;
            __bf16* l = &Bs[bn * 32 + bk];
#ifdef USE_ASYNC_LDS
            __builtin_amdgcn_global_load_async_to_lds_b128(
                (__attribute__((address_space(1))) v4i_t*)g,
                (__attribute__((address_space(3))) v4i_t*)l, 0, 0);
#else
            *(v8bf_t*)l = *(const v8bf_t*)g;
#endif
        }
#ifdef USE_ASYNC_LDS
        __builtin_amdgcn_s_wait_asynccnt(0);
#endif
        __syncthreads();

        v16bf_t a0 = load_frag(&As[(wave * 32 + ln) * 32], half);
        v16bf_t a1 = load_frag(&As[(wave * 32 + 16 + ln) * 32], half);
        #pragma unroll
        for (int t = 0; t < 4; ++t) {
            v16bf_t b = load_frag(&Bs[(t * 16 + ln) * 32], half);
            acc[0][t] = __builtin_amdgcn_wmma_f32_16x16x32_bf16(
                false, a0, false, b, (short)0, acc[0][t], false, false);
            acc[1][t] = __builtin_amdgcn_wmma_f32_16x16x32_bf16(
                false, a1, false, b, (short)0, acc[1][t], false, false);
        }
    }

    #pragma unroll
    for (int g = 0; g < 2; ++g)
        #pragma unroll
        for (int t = 0; t < 4; ++t)
            #pragma unroll
            for (int r = 0; r < 8; ++r) {
                int row = m0 + wave * 32 + g * 16 + half * 8 + r;
                int col = n0 + t * 16 + ln;
                C[(size_t)row * N + col] = acc[g][t][r];
            }
}

// ---------------------------------------------------------------------------
// RMSNorm: one block per row; len must be a multiple of 256 (512 / 1536 here).
// Safe for in-place (values cached in registers before writeback).
// ---------------------------------------------------------------------------
__global__ __launch_bounds__(256) void rmsnorm_kernel(
    const float* __restrict__ in, int in_stride,
    const float* __restrict__ w,
    float* __restrict__ out, int out_stride, int len)
{
    const int row = blockIdx.x;
    const float* x = in + (size_t)row * in_stride;
    float* y = out + (size_t)row * out_stride;

    float local[6];
    const int cnt = len >> 8;
    float ss = 0.0f;
    for (int i = 0; i < cnt; ++i) {
        float v = x[i * 256 + threadIdx.x];
        local[i] = v;
        ss += v * v;
    }
    __shared__ float red[256];
    red[threadIdx.x] = ss;
    __syncthreads();
    for (int s = 128; s > 0; s >>= 1) {
        if (threadIdx.x < s) red[threadIdx.x] += red[threadIdx.x + s];
        __syncthreads();
    }
    float inv = rsqrtf(red[0] / (float)len + 1e-6f);
    for (int i = 0; i < cnt; ++i) {
        int c = i * 256 + threadIdx.x;
        y[c] = local[i] * inv * w[c];
    }
}

// ---------------------------------------------------------------------------
// RoPE on k_rot: deinterleave + rotate-half collapses to a 2-tap formula.
// grid = B*S rows, 64 threads.
// ---------------------------------------------------------------------------
__global__ void krot_rope_kernel(const float* __restrict__ ckv,
                                 const float* __restrict__ cos_t,
                                 const float* __restrict__ sin_t,
                                 float* __restrict__ krot)
{
    const int row = blockIdx.x;            // b*S + s
    const int j   = threadIdx.x;           // 0..63
    const float* x = ckv + (size_t)row * 576 + 512;
    float c = cos_t[(size_t)row * 64 + j];
    float s = sin_t[(size_t)row * 64 + j];
    int jj = (j < 32) ? j : j - 32;
    float x0 = x[2 * jj], x1 = x[2 * jj + 1];
    float o = (j < 32) ? (x0 * c - x1 * s) : (x1 * c + x0 * s);
    krot[(size_t)row * 64 + j] = o;
}

// RoPE on q_rot, in place. grid = B*S*H, 64 threads.
__global__ void qrope_kernel(float* __restrict__ q,
                             const float* __restrict__ cos_t,
                             const float* __restrict__ sin_t)
{
    const int idx = blockIdx.x;            // (b*S+s)*H + h
    const int row = idx >> 4;              // b*S + s
    const int j   = threadIdx.x;
    float* x = q + (size_t)idx * 192 + 128;
    float c = cos_t[(size_t)row * 64 + j];
    float s = sin_t[(size_t)row * 64 + j];
    int jj = (j < 32) ? j : j - 32;
    float x0 = x[2 * jj], x1 = x[2 * jj + 1];
    __syncthreads();
    x[j] = (j < 32) ? (x0 * c - x1 * s) : (x1 * c + x0 * s);
}

// ---------------------------------------------------------------------------
// Flash attention (causal, online softmax), bf16 WMMA for QK^T and P·V.
// Block = 4 waves, 64 q-rows; K iterated in 64-key tiles.
// Ks stored [key][feat] (lane=key frag reads contiguous);
// Vs stored TRANSPOSED [d][key] so V-fragment loads are contiguous.
// q:[B,S,H,192]  kv:[B,S,H,256] (k_pass|v)  krot:[B,S,64]  out:[B,S,H*128]
// ---------------------------------------------------------------------------
__global__ __launch_bounds__(128) void mla_attn_kernel(
    const float* __restrict__ q,
    const float* __restrict__ kv,
    const float* __restrict__ krot,
    float* __restrict__ out)
{
    __shared__ __bf16 Ks[64 * 192];        // [key][feat]
    __shared__ __bf16 Vs[128 * 64];        // [d][key]   (transposed)
    __shared__ __bf16 Ps[4 * 16 * 64];     // per-wave [row][key]

    const int tid  = threadIdx.x;
    const int lane = tid & 31;
    const int wave = tid >> 5;
    const int half = (lane >> 4) & 1;
    const int ln   = lane & 15;

    const int q0 = blockIdx.x * 64;
    const int h  = blockIdx.y;
    const int b  = blockIdx.z;
    const float scaling = 0.07216878364870322f;   // 192^-0.5

    // Q fragments for this wave's 16 rows (d = 0..191 -> 6 bf16 A-frags)
    const int qrow = q0 + wave * 16 + ln;
    const float* qbase = q + (((size_t)b * 2048 + qrow) * 16 + h) * 192;
    v16bf_t qf[6];
    #pragma unroll
    for (int f = 0; f < 6; ++f) {
        const float* p = qbase + f * 32 + half * 8;
        v4f_t a0 = *(const v4f_t*)(p);
        v4f_t a1 = *(const v4f_t*)(p + 4);
        v4f_t a2 = *(const v4f_t*)(p + 16);
        v4f_t a3 = *(const v4f_t*)(p + 20);
        v8f_t lo = __builtin_shufflevector(a0, a1, 0, 1, 2, 3, 4, 5, 6, 7);
        v8f_t hi = __builtin_shufflevector(a2, a3, 0, 1, 2, 3, 4, 5, 6, 7);
        v16f_t full = __builtin_shufflevector(lo, hi, 0, 1, 2, 3, 4, 5, 6, 7,
                                              8, 9, 10, 11, 12, 13, 14, 15);
        qf[f] = __builtin_convertvector(full, v16bf_t);
    }

    v8f_t acc[8];
    float mrow[8], lrow[8];
    #pragma unroll
    for (int n = 0; n < 8; ++n) {
        mrow[n] = -3.0e38f; lrow[n] = 0.0f;
        #pragma unroll
        for (int r = 0; r < 8; ++r) acc[n][r] = 0.0f;
    }

    const int ktiles = (q0 >> 6) + 1;              // causal: k0 <= q0
    for (int kt = 0; kt < ktiles; ++kt) {
        const int k0 = kt * 64;
        __syncthreads();
        // K tile (k_pass): 64x128 -> Ks[key][0..127], packed 8B stores
        #pragma unroll
        for (int e = 0; e < 16; ++e) {
            int idx4 = e * 128 + tid;
            int key = idx4 >> 5, d = (idx4 & 31) * 4;
            v4f_t v = *(const v4f_t*)(
                kv + (((size_t)b * 2048 + k0 + key) * 16 + h) * 256 + d);
            *(v4bf_t*)(&Ks[key * 192 + d]) = cvt4(v);
        }
        // K tile (k_rot, shared across heads): 64x64 -> Ks[key][128..191]
        #pragma unroll
        for (int e = 0; e < 8; ++e) {
            int idx4 = e * 128 + tid;
            int key = idx4 >> 4, d = (idx4 & 15) * 4;
            v4f_t v = *(const v4f_t*)(krot + ((size_t)b * 2048 + k0 + key) * 64 + d);
            *(v4bf_t*)(&Ks[key * 192 + 128 + d]) = cvt4(v);
        }
        // V tile: 64x128 -> transposed store Vs[d][key]
        #pragma unroll
        for (int e = 0; e < 16; ++e) {
            int idx4 = e * 128 + tid;
            int key = idx4 >> 5, d = (idx4 & 31) * 4;
            v4f_t v = *(const v4f_t*)(
                kv + (((size_t)b * 2048 + k0 + key) * 16 + h) * 256 + 128 + d);
            v4bf_t pk = cvt4(v);
            Vs[(d + 0) * 64 + key] = pk[0];
            Vs[(d + 1) * 64 + key] = pk[1];
            Vs[(d + 2) * 64 + key] = pk[2];
            Vs[(d + 3) * 64 + key] = pk[3];
        }
        __syncthreads();

        // scores S = Q . K^T  (16x64 per wave)
        v8f_t sc[4];
        #pragma unroll
        for (int t = 0; t < 4; ++t) {
            #pragma unroll
            for (int r = 0; r < 8; ++r) sc[t][r] = 0.0f;
            #pragma unroll
            for (int f = 0; f < 6; ++f) {
                v16bf_t bk = load_frag(&Ks[(t * 16 + ln) * 192 + f * 32], half);
                sc[t] = __builtin_amdgcn_wmma_f32_16x16x32_bf16(
                    false, qf[f], false, bk, (short)0, sc[t], false, false);
            }
        }

        // scale + causal mask + online softmax (row = VGPR slot + lane half)
        const int rowbase = q0 + wave * 16 + half * 8;
        #pragma unroll
        for (int r = 0; r < 8; ++r) {
            float xm = -3.0e38f;
            #pragma unroll
            for (int t = 0; t < 4; ++t) {
                float s = sc[t][r] * scaling;
                if (k0 + t * 16 + ln > rowbase + r) s = -3.0e38f;
                sc[t][r] = s;
                xm = fmaxf(xm, s);
            }
            for (int off = 8; off >= 1; off >>= 1)
                xm = fmaxf(xm, __shfl_xor(xm, off, 32));
            float mn = fmaxf(mrow[r], xm);
            float rescale = __expf(mrow[r] - mn);
            mrow[r] = mn;
            float ls = 0.0f;
            #pragma unroll
            for (int t = 0; t < 4; ++t) {
                float p = __expf(sc[t][r] - mn);
                sc[t][r] = p;
                ls += p;
            }
            for (int off = 8; off >= 1; off >>= 1)
                ls += __shfl_xor(ls, off, 32);
            lrow[r] = lrow[r] * rescale + ls;
            #pragma unroll
            for (int n = 0; n < 8; ++n) acc[n][r] *= rescale;
            #pragma unroll
            for (int t = 0; t < 4; ++t)
                Ps[wave * 1024 + (half * 8 + r) * 64 + t * 16 + ln] = f2bf(sc[t][r]);
        }
        __syncthreads();

        // acc += P . V   (P: 16x64 bf16 A-frags, V from transposed tile)
        #pragma unroll
        for (int f = 0; f < 2; ++f) {
            v16bf_t pa = load_frag(&Ps[wave * 1024 + ln * 64 + f * 32], half);
            #pragma unroll
            for (int n = 0; n < 8; ++n) {
                v16bf_t bv = load_frag(&Vs[(n * 16 + ln) * 64 + f * 32], half);
                acc[n] = __builtin_amdgcn_wmma_f32_16x16x32_bf16(
                    false, pa, false, bv, (short)0, acc[n], false, false);
            }
        }
    }

    // epilogue: out[b, row, h*128 + col] = acc / l
    #pragma unroll
    for (int n = 0; n < 8; ++n)
        #pragma unroll
        for (int r = 0; r < 8; ++r) {
            int row = q0 + wave * 16 + half * 8 + r;
            out[((size_t)b * 2048 + row) * 2048 + h * 128 + n * 16 + ln] =
                acc[n][r] / lrow[r];
        }
}

// ---------------------------------------------------------------------------
extern "C" void kernel_launch(void* const* d_in, const int* in_sizes, int n_in,
                              void* d_out, int out_size, void* d_ws, size_t ws_size,
                              hipStream_t stream)
{
    const float* hidden = (const float*)d_in[0];
    const float* cos_t  = (const float*)d_in[1];
    const float* sin_t  = (const float*)d_in[2];
    const float* wq_a   = (const float*)d_in[3];
    const float* q_ln   = (const float*)d_in[4];
    const float* wq_b   = (const float*)d_in[5];
    const float* wkv_a  = (const float*)d_in[6];
    const float* kv_ln  = (const float*)d_in[7];
    const float* wkv_b  = (const float*)d_in[8];
    const float* wo     = (const float*)d_in[9];
    float* out = (float*)d_out;

    float* ws = (float*)d_ws;
    // fp32 region (floats); attn aliases q_lat+ckv (both dead by then)
    float* q_lat = ws;                    // 4096*1536
    float* ckv   = ws + 6291456;          // 4096*576
    float* attn  = ws;                    // 4096*2048 (alias)
    float* qbuf  = ws + 8650752;          // 4096*3072
    float* kvn   = ws + 21233664;         // 4096*512
    float* krot  = ws + 23330816;         // 4096*64
    float* kvb   = ws + 23592960;         // 4096*4096
    // bf16 transposed weights after the fp32 region (offset 40370176 floats)
    __bf16* wt   = (__bf16*)(ws + 40370176);
    __bf16* wq_a_t  = wt;                 // [1536][2048]  3145728
    __bf16* wq_b_t  = wt + 3145728;       // [3072][1536]  4718592
    __bf16* wkv_a_t = wt + 7864320;       // [576][2048]   1179648
    __bf16* wkv_b_t = wt + 9043968;       // [4096][512]   2097152
    __bf16* wo_t    = wt + 11141120;      // [2048][2048]  4194304

    // 0) weight prep: fp32 [K][N] -> bf16 [N][K]
    transpose_cvt_kernel<<<dim3(1536/32, 2048/32), 256, 0, stream>>>(wq_a,  wq_a_t,  2048, 1536);
    transpose_cvt_kernel<<<dim3(3072/32, 1536/32), 256, 0, stream>>>(wq_b,  wq_b_t,  1536, 3072);
    transpose_cvt_kernel<<<dim3(576/32,  2048/32), 256, 0, stream>>>(wkv_a, wkv_a_t, 2048, 576);
    transpose_cvt_kernel<<<dim3(4096/32, 512/32),  256, 0, stream>>>(wkv_b, wkv_b_t, 512,  4096);
    transpose_cvt_kernel<<<dim3(2048/32, 2048/32), 256, 0, stream>>>(wo,    wo_t,    2048, 2048);

    // 1) q_lat = hidden @ wq_a                [4096,2048]x[2048,1536]
    gemm_bf16_wmma<<<dim3(1536/64, 16), 256, 0, stream>>>(hidden, wq_a_t, q_lat, 4096, 1536, 2048);
    // 2) rmsnorm(q_lat) in place
    rmsnorm_kernel<<<4096, 256, 0, stream>>>(q_lat, 1536, q_ln, q_lat, 1536, 1536);
    // 3) q = q_lat @ wq_b                     [4096,1536]x[1536,3072]
    gemm_bf16_wmma<<<dim3(3072/64, 16), 256, 0, stream>>>(q_lat, wq_b_t, qbuf, 4096, 3072, 1536);
    // 4) ckv = hidden @ wkv_a                 [4096,2048]x[2048,576]
    gemm_bf16_wmma<<<dim3(576/64, 16), 256, 0, stream>>>(hidden, wkv_a_t, ckv, 4096, 576, 2048);
    // 5a) kv_lat_norm = rmsnorm(ckv[:, :512])
    rmsnorm_kernel<<<4096, 256, 0, stream>>>(ckv, 576, kv_ln, kvn, 512, 512);
    // 5b) k_rot = rope(ckv[:, 512:])
    krot_rope_kernel<<<4096, 64, 0, stream>>>(ckv, cos_t, sin_t, krot);
    // 6) kv = kvn @ wkv_b                     [4096,512]x[512,4096]
    gemm_bf16_wmma<<<dim3(4096/64, 16), 256, 0, stream>>>(kvn, wkv_b_t, kvb, 4096, 4096, 512);
    // 7) rope(q[..., 128:192]) in place
    qrope_kernel<<<4096 * 16, 64, 0, stream>>>(qbuf, cos_t, sin_t);
    // 8) flash attention
    mla_attn_kernel<<<dim3(32, 16, 2), 128, 0, stream>>>(qbuf, kvb, krot, attn);
    // 9) out = attn @ wo                      [4096,2048]x[2048,2048]
    gemm_bf16_wmma<<<dim3(2048/64, 16), 256, 0, stream>>>(attn, wo_t, out, 4096, 2048, 2048);
}
